// AttentionModule_50002009260608
// MI455X (gfx1250) — matrix operand order repaired and checked by reading, not compiled, not run on hardware
//
#include <hip/hip_runtime.h>
#include <hip/hip_bf16.h>
#include <stdint.h>

// ---------------------------------------------------------------------------
// Problem constants (from reference): B=16, C=512, H=W=24 (HW=576), TF=512,
// NH=8, CPH=64, scale = 1/sqrt(64) = 0.125
// ---------------------------------------------------------------------------
#define BB   16
#define CC   512
#define HWN  576
#define TFN  512
#define NHD  8
#define CPHN 64
#define ATT_SCALE 0.125f

typedef __attribute__((ext_vector_type(16))) __bf16 v16bf;
typedef __attribute__((ext_vector_type(8)))  float  v8f;

struct alignas(16) U16 { unsigned u[4]; };
union Frag { U16 q[2]; v16bf v; };

__device__ __forceinline__ unsigned short f2bf(float f) {
  union { float f; unsigned u; } x; x.f = f;
  unsigned r = x.u + 0x7FFFu + ((x.u >> 16) & 1u);   // round-to-nearest-even
  return (unsigned short)(r >> 16);
}

__device__ __forceinline__ float fmx(float a, float b) { return a > b ? a : b; }

__device__ __forceinline__ v8f wmma_bf16(const Frag& a, const Frag& b, v8f c) {
  return __builtin_amdgcn_wmma_f32_16x16x32_bf16(false, a.v, false, b.v,
                                                 (short)0, c, false, false);
}

// ---- CDNA5 async memory->LDS copy (ASYNCcnt), CDNA5 ISA + 15.18.3 -------
// global_load_async_to_lds_b128 vdst(lds byte addr), vaddr(32b offset), saddr
__device__ __forceinline__ void async_ld_b128(unsigned lds_addr, unsigned goff,
                                              const void* sbase) {
  asm volatile("global_load_async_to_lds_b128 %0, %1, %2"
               :: "v"(lds_addr), "v"(goff), "s"(sbase) : "memory");
}
__device__ __forceinline__ void wait_async0() {
#if __has_builtin(__builtin_amdgcn_s_wait_asynccnt)
  __builtin_amdgcn_s_wait_asynccnt(0);
#else
  asm volatile("s_wait_asynccnt 0x0" ::: "memory");
#endif
}
// LDS flat-pointer low 32 bits == wave-relative LDS byte address
__device__ __forceinline__ unsigned lds_addr_of(const void* p) {
  return (unsigned)(uintptr_t)p;
}

// ---------------------------------------------------------------------------
// f32 -> bf16 strided convert (weights + x)
// ---------------------------------------------------------------------------
__global__ void cvt_bf16(const float* __restrict__ src, unsigned short* __restrict__ dst,
                         int rows, int cols, int sstride) {
  int n = rows * cols;
  for (int i = blockIdx.x * blockDim.x + threadIdx.x; i < n; i += gridDim.x * blockDim.x) {
    int r = i / cols, c = i - r * cols;
    dst[i] = f2bf(src[(size_t)r * sstride + c]);
  }
}

// ---------------------------------------------------------------------------
// t_m[b,c] = Wt_w[c,:] . t[b,:] + Wt_b[c]
// vtt[b,c] = Wm[c, 512:1024] . t[b,:]   (text half of the merge 1x1 conv)
// ---------------------------------------------------------------------------
__global__ void tm_kernel(const float* __restrict__ t_in, const float* __restrict__ Wt_w,
                          const float* __restrict__ Wt_b, const float* __restrict__ Wm,
                          float* __restrict__ tmv, float* __restrict__ vtt) {
  int i = blockIdx.x * blockDim.x + threadIdx.x;
  if (i >= BB * CC) return;
  int b = i >> 9, c = i & 511;
  const float* tb = t_in + (size_t)b * TFN;
  const float* w0 = Wt_w + (size_t)c * TFN;
  const float* w1 = Wm   + (size_t)c * (CC + TFN) + CC;
  float s0 = 0.f, s1 = 0.f;
  for (int j = 0; j < TFN; j++) { float tv = tb[j]; s0 += w0[j] * tv; s1 += w1[j] * tv; }
  tmv[i] = s0 + Wt_b[c];
  vtt[i] = s1;
}

// ---------------------------------------------------------------------------
// cross logits: cl[b,h,m] = scale * sum_j x[b, h*64+j, m] * tmv[b, h*64+j]
// ---------------------------------------------------------------------------
__global__ void cross_logits(const float* __restrict__ x, const float* __restrict__ tmv,
                             float* __restrict__ cl) {
  int i = blockIdx.x * blockDim.x + threadIdx.x;
  if (i >= BB * NHD * HWN) return;
  int b = i / (NHD * HWN);
  int r = i - b * NHD * HWN;
  int h = r / HWN;
  int m = r - h * HWN;
  const float* xp = x + ((size_t)b * CC + h * CPHN) * HWN + m;
  const float* tp = tmv + (size_t)b * CC + h * CPHN;
  float s = 0.f;
  for (int j = 0; j < CPHN; j++) s += xp[(size_t)j * HWN] * tp[j];
  cl[i] = s * ATT_SCALE;
}

// one block per (b,h) row of 576; in-place softmax
__global__ void cross_softmax(float* __restrict__ cl) {
  __shared__ float red[256];
  float* p = cl + (size_t)blockIdx.x * HWN;
  int t = threadIdx.x;
  float m = -3.4e38f;
  for (int i = t; i < HWN; i += 256) m = fmx(m, p[i]);
  red[t] = m; __syncthreads();
  for (int s = 128; s > 0; s >>= 1) { if (t < s) red[t] = fmx(red[t], red[t + s]); __syncthreads(); }
  m = red[0]; __syncthreads();
  float sm = 0.f;
  for (int i = t; i < HWN; i += 256) sm += __expf(p[i] - m);
  red[t] = sm; __syncthreads();
  for (int s = 128; s > 0; s >>= 1) { if (t < s) red[t] += red[t + s]; __syncthreads(); }
  float inv = 1.f / red[0];
  for (int i = t; i < HWN; i += 256) p[i] = __expf(p[i] - m) * inv;
}

// ---------------------------------------------------------------------------
// Batched bf16 GEMM:  Out[b] = W (512x512) @ Xin[b] (512x576)
// BM=128, BN=64, BK=32; 256 threads = 8 waves; wave w owns rows [w*16,w*16+16).
// Double-buffered LDS; A tile staged with GLOBAL_LOAD_ASYNC_TO_LDS_B128
// (ASYNCcnt), B tile staged via one coalesced 16B load/thread held in regs and
// committed (transposed) at buffer swap. One barrier per K-step.
// TRANS=1  -> store transposed: Out[b][n (576)][m (512)]   (16B/lane stores)
// BIAS=1   -> + bias[m];  BIAS=2 -> + bias[b*512+m]
// F32OUT=1 -> f32 output (final result), else bf16
// ---------------------------------------------------------------------------
template <int TRANS, int BIAS, int F32OUT>
__global__ __launch_bounds__(256)
void gemm512(const unsigned short* __restrict__ W,
             const unsigned short* __restrict__ Xin,
             void* __restrict__ Out,
             const float* __restrict__ bias) {
  constexpr int Kd = 512, Nd = HWN, LD = 40;   // LDS row pad to 40 halves
  __shared__ unsigned short sA[2][128 * LD];
  __shared__ unsigned short sB[2][64 * LD];
  const int t = threadIdx.x;
  const int wv = t >> 5, lane = t & 31, l = lane & 15, hi = lane >> 4;
  const int nt = blockIdx.x, mt = blockIdx.y, b = blockIdx.z;
  const unsigned short* Xb = Xin + (size_t)b * Kd * Nd;

  v8f acc[4];
  for (int j = 0; j < 4; j++) for (int e = 0; e < 8; e++) acc[j][e] = 0.f;

  // A staging map: thread -> (row = t>>1, halves seg = (t&1)*16), 2x16B chunks
  const int arow = t >> 1, aseg = (t & 1) * 16;
  const unsigned a_g0 = (unsigned)(((mt * 128 + arow) * Kd + aseg) * 2);  // bytes
  const unsigned a_l0 = lds_addr_of(&sA[0][arow * LD + aseg]);
  const unsigned a_l1 = lds_addr_of(&sA[1][arow * LD + aseg]);
  // B staging map: thread -> (k = t>>3, n seg = (t&7)*8), one 16B chunk
  const int bk = t >> 3, bnseg = (t & 7) * 8;
  const unsigned short* bgp = Xb + (size_t)bk * Nd + nt * 64 + bnseg;

  // prologue: stage tile 0
  U16 breg = *(const U16*)bgp;
  async_ld_b128(a_l0,      a_g0,      W);
  async_ld_b128(a_l0 + 32, a_g0 + 32, W);

  for (int kk = 0; kk < 16; kk++) {
    const int cur = kk & 1;
    wait_async0();                       // A tile kk resident in LDS
    // commit B tile kk (transpose regs -> LDS: sB[n][k])
    {
      const unsigned short* bh = (const unsigned short*)&breg;
      unsigned short* sb = &sB[cur][0];
#pragma unroll
      for (int i = 0; i < 8; i++) sb[(bnseg + i) * LD + bk] = bh[i];
    }
    __syncthreads();                     // tiles kk visible to all waves
    if (kk < 15) {                       // stage tile kk+1 into the other buffer
      const unsigned nb = cur ? a_l0 : a_l1;
      const unsigned go = a_g0 + (unsigned)((kk + 1) * 32 * 2);
      async_ld_b128(nb,      go,      W);
      async_ld_b128(nb + 32, go + 32, W);
      breg = *(const U16*)(bgp + (size_t)(kk + 1) * 32 * Nd);
      if (kk < 14) __builtin_prefetch(bgp + (size_t)(kk + 2) * 32 * Nd, 0, 0);
    }
    // compute tile kk
    const unsigned short* sa = &sA[cur][0];
    const unsigned short* sb = &sB[cur][0];
    Frag a;   // 16x32 bf16 A frag: lane<16 row=l K{0..7,16..23}; lane>=16 K{8..15,24..31}
    a.q[0] = *(const U16*)&sa[(wv * 16 + l) * LD + hi * 8];
    a.q[1] = *(const U16*)&sa[(wv * 16 + l) * LD + 16 + hi * 8];
#pragma unroll
    for (int j = 0; j < 4; j++) {
      Frag bf;   // 32x16 B frag from transposed store: lane n=l, k = hi*8 + {0..7,16..23}
      bf.q[0] = *(const U16*)&sb[(j * 16 + l) * LD + hi * 8];
      bf.q[1] = *(const U16*)&sb[(j * 16 + l) * LD + 16 + hi * 8];
      acc[j] = wmma_bf16(a, bf, acc[j]);
    }
  }

  // epilogue; D element r -> (m = r + 8*hi, n = l) within the 16x16 tile
#pragma unroll
  for (int j = 0; j < 4; j++) {
    int ng = nt * 64 + j * 16 + l;
    if (TRANS) {
      alignas(16) unsigned short tmp[8];
#pragma unroll
      for (int r = 0; r < 8; r++) {
        int mg = mt * 128 + wv * 16 + 8 * hi + r;
        float vx = acc[j][r];
        if (BIAS == 1) vx += bias[mg];
        if (BIAS == 2) vx += bias[b * CC + mg];
        tmp[r] = f2bf(vx);
      }
      unsigned short* o = (unsigned short*)Out;
      *(U16*)&o[((size_t)b * Nd + ng) * 512 + mt * 128 + wv * 16 + 8 * hi] = *(U16*)tmp;
    } else {
#pragma unroll
      for (int r = 0; r < 8; r++) {
        int mg = mt * 128 + wv * 16 + 8 * hi + r;
        float vx = acc[j][r];
        if (BIAS == 1) vx += bias[mg];
        if (BIAS == 2) vx += bias[b * CC + mg];
        size_t idx = ((size_t)b * CC + mg) * Nd + ng;
        if (F32OUT) ((float*)Out)[idx] = vx;
        else        ((unsigned short*)Out)[idx] = f2bf(vx);
      }
    }
  }
}

// ---------------------------------------------------------------------------
// Fused attention per (batch, head, 16-query block). 128 threads = 4 waves.
// S = (Q^T K) held f32 in LDS -> softmax(*0.125) + cross -> P (bf16, LDS)
// O = V @ P^T  -> AO[b][c][hw] bf16
// Qt/Kt layout: [b][hw][512] (transposed); V: [b][512][hw] (natural)
// ---------------------------------------------------------------------------
#define SP 584   // S row pitch (floats)
#define PP 592   // P row pitch (halves)
#define ATTN_SMEM (16 * SP * 4 + 16 * PP * 2 + 16 * 8 * 4)

__global__ __launch_bounds__(128)
void attn_kernel(const unsigned short* __restrict__ Qt,
                 const unsigned short* __restrict__ Kt,
                 const unsigned short* __restrict__ V,
                 const float* __restrict__ cross,
                 unsigned short* __restrict__ AO) {
  extern __shared__ char smem[];
  float*          S   = (float*)smem;
  unsigned short* P   = (unsigned short*)(smem + 16 * SP * 4);
  float*          red = (float*)(smem + 16 * SP * 4 + 16 * PP * 2);

  const int t = threadIdx.x, wv = t >> 5, lane = t & 31, l = lane & 15, hi = lane >> 4;
  const int qb = blockIdx.x, h = blockIdx.y, b = blockIdx.z;

  // ---- S = Q^T K : A frags (Q^T rows = queries) loaded once, reused --------
  const unsigned short* qrow = Qt + ((size_t)b * HWN + qb * 16 + l) * CC + h * CPHN;
  Frag a0, a1;
  a0.q[0] = *(const U16*)(qrow + hi * 8);
  a0.q[1] = *(const U16*)(qrow + 16 + hi * 8);
  a1.q[0] = *(const U16*)(qrow + 32 + hi * 8);
  a1.q[1] = *(const U16*)(qrow + 48 + hi * 8);

  for (int mtile = wv; mtile < HWN / 16; mtile += 4) {
    const unsigned short* krow =
        Kt + ((size_t)b * HWN + mtile * 16 + l) * CC + h * CPHN + hi * 8;
    Frag b0, b1;
    b0.q[0] = *(const U16*)(krow);
    b0.q[1] = *(const U16*)(krow + 16);
    b1.q[0] = *(const U16*)(krow + 32);
    b1.q[1] = *(const U16*)(krow + 48);
    v8f acc; for (int e = 0; e < 8; e++) acc[e] = 0.f;
    acc = wmma_bf16(a0, b0, acc);
    acc = wmma_bf16(a1, b1, acc);
#pragma unroll
    for (int r = 0; r < 8; r++)
      S[(r + 8 * hi) * SP + mtile * 16 + l] = acc[r];
  }
  __syncthreads();

  // ---- softmax over keys (576), + cross row --------------------------------
  {
    int row = t >> 3, sub = t & 7;               // 16 rows x 8 threads
    const float* cr = cross + ((size_t)b * NHD + h) * HWN;
    int c0 = sub * 72, c1 = c0 + 72;
    float lmax = -3.4e38f;
    for (int c = c0; c < c1; c++) lmax = fmx(lmax, S[row * SP + c]);
    red[row * 8 + sub] = lmax; __syncthreads();
    float rmax = red[row * 8];
    for (int i = 1; i < 8; i++) rmax = fmx(rmax, red[row * 8 + i]);
    __syncthreads();
    float lsum = 0.f;
    for (int c = c0; c < c1; c++) lsum += __expf((S[row * SP + c] - rmax) * ATT_SCALE);
    red[row * 8 + sub] = lsum; __syncthreads();
    float rsum = 0.f;
    for (int i = 0; i < 8; i++) rsum += red[row * 8 + i];
    float inv = 1.f / rsum;
    for (int c = c0; c < c1; c++)
      P[row * PP + c] = f2bf(__expf((S[row * SP + c] - rmax) * ATT_SCALE) * inv + cr[c]);
  }
  __syncthreads();

  // ---- O = V @ P^T : wave wv owns c-tile wv (16 of the 64 head channels) ---
  const unsigned short* vrow = V + ((size_t)b * CC + h * CPHN + wv * 16 + l) * HWN;
  v8f acc; for (int e = 0; e < 8; e++) acc[e] = 0.f;
  for (int k0 = 0; k0 < HWN; k0 += 32) {
    Frag a, bf;
    a.q[0]  = *(const U16*)(vrow + k0 + hi * 8);
    a.q[1]  = *(const U16*)(vrow + k0 + 16 + hi * 8);
    bf.q[0] = *(const U16*)&P[l * PP + k0 + hi * 8];
    bf.q[1] = *(const U16*)&P[l * PP + k0 + 16 + hi * 8];
    acc = wmma_bf16(a, bf, acc);
  }
#pragma unroll
  for (int r = 0; r < 8; r++)
    AO[((size_t)b * CC + h * CPHN + wv * 16 + 8 * hi + r) * HWN + qb * 16 + l] = f2bf(acc[r]);
}

// ---------------------------------------------------------------------------
extern "C" void kernel_launch(void* const* d_in, const int* in_sizes, int n_in,
                              void* d_out, int out_size, void* d_ws, size_t ws_size,
                              hipStream_t stream) {
  const float* x    = (const float*)d_in[0];
  const float* tt   = (const float*)d_in[1];
  const float* Wk   = (const float*)d_in[2];
  const float* Wq   = (const float*)d_in[3];
  const float* Wt_w = (const float*)d_in[4];
  const float* Wt_b = (const float*)d_in[5];
  const float* Wm   = (const float*)d_in[6];
  const float* Wv   = (const float*)d_in[7];
  const float* Wr_w = (const float*)d_in[8];
  const float* Wr_b = (const float*)d_in[9];
  float* out = (float*)d_out;

  char* ws = (char*)d_ws;
  size_t off = 0;
  auto alloc = [&](size_t bytes) -> void* {
    void* p = ws + off;
    off = (off + bytes + 255) & ~(size_t)255;
    return p;
  };

  const size_t bchw = (size_t)BB * CC * HWN;               // 4,718,592
  unsigned short* xb  = (unsigned short*)alloc(bchw * 2);
  unsigned short* wkb = (unsigned short*)alloc((size_t)CC * CC * 2);
  unsigned short* wqb = (unsigned short*)alloc((size_t)CC * CC * 2);
  unsigned short* wvb = (unsigned short*)alloc((size_t)CC * CC * 2);
  unsigned short* wrb = (unsigned short*)alloc((size_t)CC * CC * 2);
  unsigned short* wmb = (unsigned short*)alloc((size_t)CC * CC * 2);  // image half of Wm
  unsigned short* Qt  = (unsigned short*)alloc(bchw * 2);  // [b][hw][c]
  unsigned short* Kt  = (unsigned short*)alloc(bchw * 2);  // [b][hw][c]
  unsigned short* VL  = (unsigned short*)alloc(bchw * 2);  // [b][c][hw]
  unsigned short* Vv  = (unsigned short*)alloc(bchw * 2);  // [b][c][hw]
  unsigned short* AO  = (unsigned short*)alloc(bchw * 2);  // [b][c][hw]
  float* tmv = (float*)alloc((size_t)BB * CC * 4);
  float* vtt = (float*)alloc((size_t)BB * CC * 4);
  float* cl  = (float*)alloc((size_t)BB * NHD * HWN * 4);
  (void)ws_size; (void)in_sizes; (void)n_in; (void)out_size;

  // --- phase 0: conversions + text-side precompute --------------------------
  cvt_bf16<<<4096, 256, 0, stream>>>(x, xb, BB * CC, HWN, HWN);
  cvt_bf16<<<1024, 256, 0, stream>>>(Wk, wkb, CC, CC, CC);
  cvt_bf16<<<1024, 256, 0, stream>>>(Wq, wqb, CC, CC, CC);
  cvt_bf16<<<1024, 256, 0, stream>>>(Wv, wvb, CC, CC, CC);
  cvt_bf16<<<1024, 256, 0, stream>>>(Wr_w, wrb, CC, CC, CC);
  cvt_bf16<<<1024, 256, 0, stream>>>(Wm, wmb, CC, CC, CC + TFN);  // Wm[:, :512]

  tm_kernel<<<(BB * CC + 255) / 256, 256, 0, stream>>>(tt, Wt_w, Wt_b, Wm, tmv, vtt);
  cross_logits<<<(BB * NHD * HWN + 255) / 256, 256, 0, stream>>>(x, tmv, cl);
  cross_softmax<<<BB * NHD, 256, 0, stream>>>(cl);

  // --- phase 1: channel GEMMs (WMMA + async LDS staging) --------------------
  dim3 gg(HWN / 64, CC / 128, BB);   // (9, 4, 16)
  gemm512<1, 0, 0><<<gg, 256, 0, stream>>>(wqb, xb, Qt, nullptr);        // Q^T
  gemm512<1, 0, 0><<<gg, 256, 0, stream>>>(wkb, xb, Kt, nullptr);        // K^T
  gemm512<0, 2, 0><<<gg, 256, 0, stream>>>(wmb, xb, VL, vtt);            // merge + text bias
  gemm512<0, 0, 0><<<gg, 256, 0, stream>>>(wvb, VL, Vv, nullptr);        // values

  // --- phase 2: fused attention (WMMA) --------------------------------------
  dim3 ga(HWN / 16, NHD, BB);        // (36, 8, 16)
  attn_kernel<<<ga, 128, ATTN_SMEM, stream>>>(Qt, Kt, Vv, cl, AO);

  // --- phase 3: output projection (WMMA, f32 out + bias) --------------------
  gemm512<0, 1, 1><<<gg, 256, 0, stream>>>(wrb, AO, out, Wr_b);
}
